// GNNLSTM_61529701483150
// MI455X (gfx1250) — compile-verified, matrix operand order checked by reading, not run on hardware
//
#include <hip/hip_runtime.h>
#include <hip/hip_bf16.h>
#include <math.h>

// ---- problem constants ----
#define BB 8
#define TT 8
#define NN 1024
#define CCH 64
#define HH 64
#define FOUR_H 256
#define KDIM 128           // C + H
#define ALPHA_LR 0.2f
#define NEGV -9.0e15f
#define ROWS (BB*NN)       // 8192

typedef __attribute__((ext_vector_type(16))) __bf16 v16bf;
typedef __attribute__((ext_vector_type(4)))  __bf16 v4bf;
typedef __attribute__((ext_vector_type(8)))  float  v8f;
typedef __attribute__((ext_vector_type(4)))  float  v4f;
typedef __attribute__((ext_vector_type(4)))  unsigned int v4u;

__device__ __forceinline__ unsigned short f2bf(float f) {
  union { __bf16 b; unsigned short u; } c;
  c.b = (__bf16)f;                       // hardware v_cvt bf16
  return c.u;
}
__device__ __forceinline__ float bf2f(unsigned short b) {
  union { unsigned u; float f; } c; c.u = ((unsigned)b) << 16;
  return c.f;
}
__device__ __forceinline__ float sigm(float x) { return 1.0f / (1.0f + __expf(-x)); }
__device__ __forceinline__ float eluf(float x) { return x > 0.0f ? x : (__expf(x) - 1.0f); }

// 16-bit WMMA fragment (ISA 7.12.2): lane holds M(or N)=lane%16; its 16 bf16
// elements are two K-contiguous groups of 8 starting at K = half and K = 16+half,
// where half = 8*(lane>=16).  => two 16-byte loads per fragment when K is
// contiguous in memory.
union Frag {
  v16bf v;
  v4u   q[2];
  v4bf  b4[4];
  unsigned short u[16];
};

// ---------------- init: h = x0@c_init_w + c_init_b ; c = x0@h_init_w + h_init_b ----------------
__global__ void init_hc(const float* __restrict__ node_fea,
                        const float* __restrict__ h_init_w, const float* __restrict__ h_init_b,
                        const float* __restrict__ c_init_w, const float* __restrict__ c_init_b,
                        float* __restrict__ h, float* __restrict__ c) {
  int idx = blockIdx.x * blockDim.x + threadIdx.x;   // ROWS*HH
  int col = idx & 63;
  int row = idx >> 6;
  int b = row >> 10, n = row & 1023;
  const float* x0 = node_fea + (((size_t)b * TT + 0) * NN + n) * CCH;
  float accH = c_init_b[col], accC = h_init_b[col];
  #pragma unroll 4
  for (int k = 0; k < CCH; k += 4) {
    v4f x = *(const v4f*)(x0 + k);
    #pragma unroll
    for (int i = 0; i < 4; ++i) {
      accC += x[i] * h_init_w[(k + i) * HH + col];
      accH += x[i] * c_init_w[(k + i) * HH + col];
    }
  }
  c[idx] = accC;
  h[idx] = accH;
}

// ---------------- misc init: W -> bf16 transposed (col-major, K contiguous), zero loss ----------------
__global__ void init_misc(const float* __restrict__ W, unsigned short* __restrict__ WbfT,
                          float* __restrict__ loss) {
  int i = blockIdx.x * blockDim.x + threadIdx.x;     // over KDIM*FOUR_H
  if (i < KDIM * FOUR_H) {
    int k = i >> 8, col = i & 255;                   // W is (128,256) row-major
    WbfT[(size_t)col * KDIM + k] = f2bf(W[i]);
  }
  if (i == 0) *loss = 0.0f;
}

// ---------------- WhT = ([x_t, h] @ W)^T  (bf16 WMMA, one wave per 16x16 tile) ----------------
// output stored transposed per batch: WhT (BB, 256, NN) so attention B-fragments are K-contiguous.
// Two independent accumulator chains (K split) so WMMAs can dual-issue without RAW stalls.
__global__ __launch_bounds__(128) void wh_gemm(
    const float* __restrict__ node_fea, int t,
    const float* __restrict__ h,
    const unsigned short* __restrict__ WbfT,   // (256 cols, 128 k)
    unsigned short* __restrict__ WhT)          // (BB, 256, NN)
{
  int gw   = blockIdx.x * (blockDim.x >> 5) + (threadIdx.x >> 5); // 8192 waves
  int lane = threadIdx.x & 31;
  int coltile = gw & 15;
  int rowtile = gw >> 4;
  int m    = lane & 15;
  int half = (lane & 16) ? 8 : 0;
  int row  = rowtile * 16 + m;                  // A-fragment row (M = lane%16)
  int b    = row >> 10, n = row & 1023;
  int col  = coltile * 16 + m;                  // B-fragment column
  const float* xrow = node_fea + (((size_t)b * TT + t) * NN + n) * CCH;
  const float* hrow = h + (size_t)row * HH;
  const unsigned short* wcol = WbfT + (size_t)col * KDIM;

  v8f acc0 = {}, acc1 = {};
  #pragma unroll
  for (int kc = 0; kc < 4; ++kc) {
    int k0 = kc * 32;
    Frag A, Bm;
    // A: two groups of 8 consecutive k, [k0+half, +8) and [k0+16+half, +8);
    // each group lies fully inside x (k<64) or h (k>=64).
    #pragma unroll
    for (int g = 0; g < 2; ++g) {
      int ka = k0 + 16 * g + half;
      const float* src = (ka < 64) ? (xrow + ka) : (hrow + ka - 64);
      v4f f0 = *(const v4f*)(src);
      v4f f1 = *(const v4f*)(src + 4);
      A.b4[2 * g]     = __builtin_convertvector(f0, v4bf);  // v_cvt_pk_bf16_f32
      A.b4[2 * g + 1] = __builtin_convertvector(f1, v4bf);
      Bm.q[g] = *(const v4u*)(wcol + ka);       // K-contiguous: one b128 per group
    }
    if (kc & 1)
      acc1 = __builtin_amdgcn_wmma_f32_16x16x32_bf16(false, A.v, false, Bm.v,
                                                     (short)0, acc1, false, false);
    else
      acc0 = __builtin_amdgcn_wmma_f32_16x16x32_bf16(false, A.v, false, Bm.v,
                                                     (short)0, acc0, false, false);
  }
  v8f accf = acc0 + acc1;
  // D layout: lane's rows are half+0..half+7 (consecutive n) -> single b128 store transposed
  union { v4u q; v4bf b[2]; } st;
  v4f lo = { accf[0], accf[1], accf[2], accf[3] };
  v4f hi = { accf[4], accf[5], accf[6], accf[7] };
  st.b[0] = __builtin_convertvector(lo, v4bf);
  st.b[1] = __builtin_convertvector(hi, v4bf);
  int ob  = (rowtile * 16) >> 10;
  int on0 = (rowtile * 16) & 1023;
  *(v4u*)(WhT + ((size_t)ob * FOUR_H + col) * NN + on0 + half) = st.q;
}

// ---------------- s1 = Wh@a1, s2 = Wh@a2  (thread per node, coalesced over WhT) ----------------
__global__ void s_reduce(const unsigned short* __restrict__ WhT,
                         const float* __restrict__ a,     // (512,1)
                         float* __restrict__ s1, float* __restrict__ s2) {
  int idx = blockIdx.x * blockDim.x + threadIdx.x;   // 0..8191 = b*NN + n
  int b = idx >> 10, n = idx & 1023;
  const unsigned short* base = WhT + (size_t)b * FOUR_H * NN + n;
  float p1 = 0.0f, p2 = 0.0f;
  #pragma unroll 4
  for (int cidx = 0; cidx < FOUR_H; ++cidx) {
    float w = bf2f(base[(size_t)cidx * NN]);     // consecutive threads: consecutive n
    p1 += w * a[cidx];
    p2 += w * a[FOUR_H + cidx];
  }
  s1[idx] = p1;
  s2[idx] = p2;
}

// ---------------- fused masked-softmax + (att @ Wh) WMMA + ELU + LSTM update ----------------
// one workgroup = (batch b, 16-row tile). 4 waves; wave w owns h-columns 16w+64q, q=0..3,
// so each wave holds the i/f/o/g gates for its 16 h-columns.
__global__ __launch_bounds__(128) void attn_lstm(
    const float* __restrict__ adj,
    const float* __restrict__ s1, const float* __restrict__ s2,
    const unsigned short* __restrict__ WhT,
    float* __restrict__ h, float* __restrict__ c)
{
  __shared__ __align__(16) unsigned short pLds[16 * NN]; // exp(e-rowmax), bf16 (32 KB)
  __shared__ float red[16][8];
  __shared__ float rowmax[16];
  __shared__ float rowinv[16];

  int wg  = blockIdx.x;
  int b   = wg >> 6;                 // 8 batches * 64 row tiles
  int n0  = (wg & 63) * 16;
  int tid = threadIdx.x;
  int r   = tid >> 3;                // 0..15 (row within tile)
  int sub = tid & 7;                 // 0..7

  const float* s1b = s1 + b * NN;
  const float* s2b = s2 + b * NN;

  // phase 1: row max of masked leaky-relu logits (float4 per iteration)
  {
    int i = n0 + r;
    float s1v = s1b[i];
    const float* adjrow = adj + (size_t)i * NN;
    float mx = -3.0e38f;
    for (int it = 0; it < 32; ++it) {
      int jj = sub * 4 + 32 * it;
      v4f av = *(const v4f*)(adjrow + jj);
      v4f sv = *(const v4f*)(s2b + jj);
      #pragma unroll
      for (int i4 = 0; i4 < 4; ++i4) {
        float x = s1v + sv[i4];
        float e = (av[i4] > 0.0f) ? ((x > 0.0f) ? x : ALPHA_LR * x) : NEGV;
        mx = fmaxf(mx, e);
      }
    }
    red[r][sub] = mx;
  }
  __syncthreads();
  if (sub == 0) {
    float mx = red[r][0];
    #pragma unroll
    for (int q = 1; q < 8; ++q) mx = fmaxf(mx, red[r][q]);
    rowmax[r] = mx;
  }
  __syncthreads();

  // phase 2: p = exp(e - max) -> LDS (bf16, b64 stores) + row sums
  {
    int i = n0 + r;
    float s1v = s1b[i];
    float mx  = rowmax[r];
    const float* adjrow = adj + (size_t)i * NN;
    float sum = 0.0f;
    for (int it = 0; it < 32; ++it) {
      int jj = sub * 4 + 32 * it;
      v4f av = *(const v4f*)(adjrow + jj);
      v4f sv = *(const v4f*)(s2b + jj);
      v4f p;
      #pragma unroll
      for (int i4 = 0; i4 < 4; ++i4) {
        float x = s1v + sv[i4];
        float e = (x > 0.0f) ? x : ALPHA_LR * x;
        p[i4] = (av[i4] > 0.0f) ? __expf(e - mx) : 0.0f;
        sum += p[i4];
      }
      *(v4bf*)(pLds + r * NN + jj) = __builtin_convertvector(p, v4bf);
    }
    red[r][sub] = sum;
  }
  __syncthreads();
  if (sub == 0) {
    float sm = red[r][0];
    #pragma unroll
    for (int q = 1; q < 8; ++q) sm += red[r][q];
    rowinv[r] = 1.0f / sm;
  }
  __syncthreads();

  // phase 3: hp_tile = p_tile @ Wh via bf16 WMMA.
  // A: 2x ds_load_b128 from LDS prob tile; B: 2x global_load_b128 from K-contiguous WhT.
  // 4 independent accumulator chains (q = i/f/o/g gate blocks) hide WMMA latency.
  int lane = tid & 31;
  int w    = tid >> 5;
  int m    = lane & 15;
  int half = (lane & 16) ? 8 : 0;

  v8f acc[4] = { {}, {}, {}, {} };
  const unsigned short* whtb = WhT + (size_t)b * FOUR_H * NN;

  for (int jc = 0; jc < 32; ++jc) {
    int j0 = jc * 32;
    Frag A;
    const unsigned short* pl = pLds + m * NN + j0;
    A.q[0] = *(const v4u*)(pl + half);
    A.q[1] = *(const v4u*)(pl + 16 + half);
    #pragma unroll
    for (int q = 0; q < 4; ++q) {
      int col = q * 64 + w * 16 + m;
      const unsigned short* wp = whtb + (size_t)col * NN + j0;
      Frag Bm;
      Bm.q[0] = *(const v4u*)(wp + half);
      Bm.q[1] = *(const v4u*)(wp + 16 + half);
      __builtin_prefetch((const void*)(wp + 32), 0, 1);   // next j-panel -> global_prefetch_b8
      acc[q] = __builtin_amdgcn_wmma_f32_16x16x32_bf16(false, A.v, false, Bm.v,
                                                       (short)0, acc[q], false, false);
    }
  }

  // epilogue: normalize rows, ELU, LSTM gate update
  int hcol = w * 16 + m;
  #pragma unroll
  for (int r8 = 0; r8 < 8; ++r8) {
    int mr = half + r8;                 // D layout row
    int n  = n0 + mr;
    float inv = rowinv[mr];
    size_t idx = ((size_t)b * NN + n) * HH + hcol;
    float iv = eluf(acc[0][r8] * inv);
    float fv = eluf(acc[1][r8] * inv);
    float ov = eluf(acc[2][r8] * inv);
    float gv = eluf(acc[3][r8] * inv);
    float cn = sigm(fv) * c[idx] + sigm(iv) * tanhf(gv);
    float hn = sigm(ov) * tanhf(cn);
    c[idx] = cn;
    h[idx] = hn;
  }
}

// ---------------- decode: pred = h@dec_w + dec_b ; loss accumulate OR final output ----------------
__global__ void decode(const float* __restrict__ h,
                       const float* __restrict__ dec_w, const float* __restrict__ dec_b,
                       const float* __restrict__ node_fea, int t_next,   // loss mode
                       float* __restrict__ out,                          // out mode when != null
                       float* __restrict__ loss) {
  __shared__ float rbuf[256];
  int idx = blockIdx.x * blockDim.x + threadIdx.x;    // ROWS*CCH
  int cc  = idx & 63;
  int row = idx >> 6;
  const float* hrow = h + (size_t)row * HH;
  float pred = dec_b[cc];
  #pragma unroll 4
  for (int k = 0; k < HH; k += 4) {
    v4f hv = *(const v4f*)(hrow + k);
    #pragma unroll
    for (int i = 0; i < 4; ++i) pred += hv[i] * dec_w[(k + i) * CCH + cc];
  }

  if (out != nullptr) {
    out[idx] = pred;
  } else {
    int b = row >> 10, n = row & 1023;
    float tgt = node_fea[(((size_t)b * TT + t_next) * NN + n) * CCH + cc];
    float d = pred - tgt;
    rbuf[threadIdx.x] = d * d;
    __syncthreads();
    for (int s = 128; s; s >>= 1) {
      if (threadIdx.x < s) rbuf[threadIdx.x] += rbuf[threadIdx.x + s];
      __syncthreads();
    }
    if (threadIdx.x == 0) atomicAdd(loss, rbuf[0] * (1.0f / (float)(ROWS * CCH)));
  }
}

extern "C" void kernel_launch(void* const* d_in, const int* in_sizes, int n_in,
                              void* d_out, int out_size, void* d_ws, size_t ws_size,
                              hipStream_t stream) {
  (void)in_sizes; (void)n_in; (void)out_size; (void)ws_size;
  const float* node_fea = (const float*)d_in[0];
  const float* adj      = (const float*)d_in[1];
  const float* W        = (const float*)d_in[2];
  const float* a        = (const float*)d_in[3];
  const float* h_init_w = (const float*)d_in[4];
  const float* h_init_b = (const float*)d_in[5];
  const float* c_init_w = (const float*)d_in[6];
  const float* c_init_b = (const float*)d_in[7];
  const float* dec_w    = (const float*)d_in[8];
  const float* dec_b    = (const float*)d_in[9];

  float* out  = (float*)d_out;            // (B,1,N,C) flat
  float* loss = out + (size_t)BB * NN * CCH;

  char* ws = (char*)d_ws;
  float* h = (float*)ws;                     ws += (size_t)ROWS * HH * 4;        // 2 MB
  float* c = (float*)ws;                     ws += (size_t)ROWS * HH * 4;        // 2 MB
  unsigned short* WhT = (unsigned short*)ws; ws += (size_t)BB * FOUR_H * NN * 2; // 4 MB
  float* s1 = (float*)ws;                    ws += (size_t)BB * NN * 4;
  float* s2 = (float*)ws;                    ws += (size_t)BB * NN * 4;
  unsigned short* WbfT = (unsigned short*)ws; ws += (size_t)KDIM * FOUR_H * 2;

  init_hc<<<(ROWS * HH) / 256, 256, 0, stream>>>(node_fea, h_init_w, h_init_b,
                                                 c_init_w, c_init_b, h, c);
  init_misc<<<(KDIM * FOUR_H + 255) / 256, 256, 0, stream>>>(W, WbfT, loss);

  for (int t = 0; t < TT; ++t) {
    wh_gemm<<<2048, 128, 0, stream>>>(node_fea, t, h, WbfT, WhT);      // 8192 waves
    s_reduce<<<ROWS / 256, 256, 0, stream>>>(WhT, a, s1, s2);
    attn_lstm<<<512, 128, 0, stream>>>(adj, s1, s2, WhT, h, c);
    if (t < TT - 1) {
      decode<<<(ROWS * CCH) / 256, 256, 0, stream>>>(h, dec_w, dec_b,
                                                     node_fea, t + 1, nullptr, loss);
    } else {
      decode<<<(ROWS * CCH) / 256, 256, 0, stream>>>(h, dec_w, dec_b,
                                                     nullptr, 0, out, loss);
    }
  }
}
